// HyperGNNBinaryAggregateLayer_15951508537735
// MI455X (gfx1250) — matrix operand order repaired and checked by reading, not compiled, not run on hardware
//
#include <hip/hip_runtime.h>
#include <hip/hip_bf16.h>
#include <math.h>

typedef __attribute__((ext_vector_type(16))) __bf16 v16bf;
typedef __attribute__((ext_vector_type(8)))  float  v8f;

#define NE_   48
#define DM_   64
#define DR_   64
#define KDIM  128
#define ROWS  96
#define BSTR  80   // padded bf16 row stride for W_c in LDS (160B rows, 32B aligned frags)

__device__ __forceinline__ size_t idxF(int b, int i, int j, int k) {
  return ((((size_t)b * NE_ + i) * NE_ + j) * NE_ + k) * DM_;
}

// Load 8 consecutive f32 from global, round to bf16, place into A-fragment
// elements [base, base+8) (ISA 16-bit A layout: those are 8 consecutive K).
__device__ __forceinline__ void load8_bf16(const float* __restrict__ p, v16bf& a, int base) {
  float4 x = *(const float4*)p;
  float4 y = *(const float4*)(p + 4);
  a[base + 0] = (__bf16)x.x; a[base + 1] = (__bf16)x.y;
  a[base + 2] = (__bf16)x.z; a[base + 3] = (__bf16)x.w;
  a[base + 4] = (__bf16)y.x; a[base + 5] = (__bf16)y.y;
  a[base + 6] = (__bf16)y.z; a[base + 7] = (__bf16)y.w;
}

__global__ __launch_bounds__(192)
void hypergnn_sib_kernel(const float* __restrict__ ht, const float* __restrict__ factor,
                         const float* __restrict__ W_c, const float* __restrict__ b_c,
                         const float* __restrict__ v_w, const float* __restrict__ W_fc,
                         const float* __restrict__ b_fc, const float* __restrict__ ln_g,
                         const float* __restrict__ ln_b, const int* __restrict__ ent_numbers,
                         float* __restrict__ out) {
  __shared__ __bf16 sWc[KDIM * BSTR];   // 20480 B
  __shared__ float  sEnergy[ROWS];
  __shared__ float  sAttn[ROWS];
  __shared__ float  sRed[8];
  __shared__ float  sOut[DM_];
  __shared__ float  sY[DM_];

  const int tid  = threadIdx.x;
  const int wave = tid >> 5;
  const int lane = tid & 31;
  const int b = blockIdx.y;
  const int i = blockIdx.x / NE_;
  const int j = blockIdx.x % NE_;

  // ---- stage W_c (128x64 f32) as bf16 into padded LDS ----
  for (int idx = tid; idx < KDIM * DM_; idx += 192) {
    int kk = idx >> 6, n = idx & 63;
    sWc[kk * BSTR + n] = (__bf16)W_c[idx];
  }

  const float* htrow = ht + (((size_t)b * NE_ + i) * NE_ + j) * DR_;

  // ---- build A fragments (16x32 bf16 per K-step) directly in registers ----
  const int half = lane >> 4;       // lane group: K sub-block
  const int mloc = lane & 15;       // row within tile
  const int m    = wave * 16 + mloc;   // global comb row 0..95
  const float* frow = (m < NE_) ? (factor + idxF(b, i, j, m))
                                : (factor + idxF(b, i, m - NE_, j));
  v16bf afrag[4];
#pragma unroll
  for (int ks = 0; ks < 4; ++ks) {
    int k0 = 32 * ks + 8 * half;    // elements 0..7 hold K = k0..k0+7
    int k1 = k0 + 16;               // elements 8..15 hold K = k1..k1+7
    const float* p0 = (k0 < DR_) ? (htrow + k0) : (frow + (k0 - DR_));
    const float* p1 = (k1 < DR_) ? (htrow + k1) : (frow + (k1 - DR_));
    load8_bf16(p0, afrag[ks], 0);
    load8_bf16(p1, afrag[ks], 8);
  }

  __syncthreads();

  // ---- 96x128 @ 128x64 GEMM: per wave 4 N-tiles x 4 K-steps of WMMA ----
  v8f acc[4];
  const v8f vzero = {0.f, 0.f, 0.f, 0.f, 0.f, 0.f, 0.f, 0.f};
#pragma unroll
  for (int t = 0; t < 4; ++t) acc[t] = vzero;
#pragma unroll
  for (int t = 0; t < 4; ++t) {
#pragma unroll
    for (int ks = 0; ks < 4; ++ks) {
      // B fragment: lane = N column (16t+lane%16), elements = K in table order
      v16bf bfrag;
#pragma unroll
      for (int s = 0; s < 16; ++s)
        bfrag[s] = sWc[(32 * ks + 8 * half + (s < 8 ? s : 8 + s)) * BSTR + 16 * t + mloc];
      acc[t] = __builtin_amdgcn_wmma_f32_16x16x32_bf16(
          false, afrag[ks], false, bfrag, (short)0, acc[t], false, false);
    }
  }

  // ---- bias + exact GELU + energy = h . v_w (stay in registers) ----
  float e[8];
#pragma unroll
  for (int r = 0; r < 8; ++r) e[r] = 0.f;
#pragma unroll
  for (int t = 0; t < 4; ++t) {
    float bc = b_c[16 * t + mloc];
    float vw = v_w[16 * t + mloc];
#pragma unroll
    for (int r = 0; r < 8; ++r) {
      float x = acc[t][r] + bc;
      float g = 0.5f * x * (1.0f + erff(x * 0.70710678118654752f));
      e[r] += g * vw;
    }
  }
  // reduce over the 16 lanes holding one row (C/D layout: lanes 0-15 rows r, 16-31 rows 8+r)
#pragma unroll
  for (int r = 0; r < 8; ++r) {
    e[r] += __shfl_xor(e[r], 1, 32);
    e[r] += __shfl_xor(e[r], 2, 32);
    e[r] += __shfl_xor(e[r], 4, 32);
    e[r] += __shfl_xor(e[r], 8, 32);
  }
  if (mloc == 0) {
#pragma unroll
    for (int r = 0; r < 8; ++r)
      sEnergy[wave * 16 + half * 8 + r] = e[r];
  }
  __syncthreads();

  // ---- masked softmax over the 96 energies ----
  const int en = ent_numbers[b];
  const bool vij = (i < en) && (j < en);
  float ev = -INFINITY;
  if (tid < ROWS) {
    int k = (tid < NE_) ? tid : (tid - NE_);
    bool maskb = (j == k) || !(vij && (k < en));
    ev = maskb ? -10000.0f : sEnergy[tid];
  }
  float mv = ev;
  for (int o = 16; o; o >>= 1) mv = fmaxf(mv, __shfl_xor(mv, o, 32));
  if (lane == 0) sRed[wave] = mv;
  __syncthreads();
  mv = sRed[0];
#pragma unroll
  for (int w = 1; w < 6; ++w) mv = fmaxf(mv, sRed[w]);
  float p = (tid < ROWS) ? __expf(ev - mv) : 0.f;
  float sv = p;
  for (int o = 16; o; o >>= 1) sv += __shfl_xor(sv, o, 32);
  __syncthreads();
  if (lane == 0) sRed[wave] = sv;
  __syncthreads();
  float ssum = 0.f;
#pragma unroll
  for (int w = 0; w < 6; ++w) ssum += sRed[w];
  if (tid < ROWS) sAttn[tid] = p / ssum;
  __syncthreads();

  // ---- attention-weighted sum of total_h (L2-resident re-read, coalesced over d) ----
  if (tid < DM_) {
    float a0 = 0.f;
    const float* f0 = factor + idxF(b, i, j, 0) + tid;
    for (int k = 0; k < NE_; ++k) a0 += sAttn[k] * f0[(size_t)k * DM_];
    for (int k = 0; k < NE_; ++k) a0 += sAttn[NE_ + k] * factor[idxF(b, i, k, j) + tid];
    sOut[tid] = a0;
  }
  __syncthreads();

  // ---- fc + bias + residual ----
  if (tid < DM_) {
    float y = b_fc[tid] + htrow[tid];
    for (int dd = 0; dd < DM_; ++dd) y += sOut[dd] * W_fc[dd * DR_ + tid];
    sY[tid] = y;
  }
  __syncthreads();

  // ---- LayerNorm over 64 ----
  if (tid < DM_) {
    float mu = 0.f;
    for (int dd = 0; dd < DM_; ++dd) mu += sY[dd];
    mu *= (1.0f / DM_);
    float var = 0.f;
    for (int dd = 0; dd < DM_; ++dd) { float dl = sY[dd] - mu; var += dl * dl; }
    var *= (1.0f / DM_);
    out[(((size_t)b * NE_ + i) * NE_ + j) * DR_ + tid] =
        ln_g[tid] * (sY[tid] - mu) * rsqrtf(var + 1e-6f) + ln_b[tid];
  }
}

extern "C" void kernel_launch(void* const* d_in, const int* in_sizes, int n_in,
                              void* d_out, int out_size, void* d_ws, size_t ws_size,
                              hipStream_t stream) {
  (void)n_in; (void)out_size; (void)d_ws; (void)ws_size;
  const float* ht    = (const float*)d_in[0];
  const float* fac   = (const float*)d_in[1];
  const float* W_c   = (const float*)d_in[2];
  const float* b_c   = (const float*)d_in[3];
  const float* v_w   = (const float*)d_in[4];
  const float* W_fc  = (const float*)d_in[5];
  const float* b_fc  = (const float*)d_in[6];
  const float* ln_g  = (const float*)d_in[7];
  const float* ln_b  = (const float*)d_in[8];
  const int*   entn  = (const int*)d_in[9];
  const int bs = in_sizes[9];               // ent_numbers has BS elements
  dim3 grid(NE_ * NE_, bs);
  hypergnn_sib_kernel<<<grid, 192, 0, stream>>>(ht, fac, W_c, b_c, v_w, W_fc,
                                                b_fc, ln_g, ln_b, entn,
                                                (float*)d_out);
}
